// LinearAttention_37048387895676
// MI455X (gfx1250) — matrix operand order repaired and checked by reading, not compiled, not run on hardware
//
#include <hip/hip_runtime.h>
#include <hip/hip_bf16.h>
#include <math.h>

// ---------------------------------------------------------------------------
// LinearAttention for MI455X (gfx1250), wave32 + v_wmma_f32_16x16x32_f16.
// Bandwidth-bound (~19 GFLOP vs ~225 MB HBM traffic after f16 intermediates):
//   - qkv intermediate stored f16, token-major [b][n][row] (vector st/ld)
//   - single-pass online k-softmax stats (k read once)
//   - context via 512 chunked blocks + f32 atomics (chip-filling)
//   - fully fused tail: q-softmax -> WMMA ctx^T@q -> WMMA w_out@attn
//     -> bias -> channel RMS-norm -> single output store
// ---------------------------------------------------------------------------

typedef __attribute__((ext_vector_type(16))) _Float16 v16h;
typedef __attribute__((ext_vector_type(8)))  _Float16 v8h;
typedef __attribute__((ext_vector_type(8)))  float    v8f;

constexpr int B_  = 16;
constexpr int C_  = 256;
constexpr int N_  = 64 * 64;   // H*W
constexpr int NH_ = 4;
constexpr int HD_ = 32;
constexpr int NM_ = 4;
constexpr int HID_ = NH_ * HD_;   // 128
constexpr int QR_ = HID_ * 3;     // 384 qkv rows

__device__ __forceinline__ v8f vzero8() {
  v8f z = {0.f, 0.f, 0.f, 0.f, 0.f, 0.f, 0.f, 0.f};
  return z;
}

// A-matrix 16-bit fragment mapping (contiguous k, matching B): lane m in 0..15 holds
// K = {0..15}; lanes 16..31 hold row m=lane-16, K = {16..31}.
__device__ __forceinline__ void afrag_mk(int lane, int j, int& m, int& k) {
  m = lane & 15;
  k = ((lane < 16) ? 0 : 16) + j;
}

// ---------------------------------------------------------------------------
// K0: pre-swizzle w_qkv (384x256) and w_out (256x128) into f16 WMMA
// A-fragment order; spare threads zero the ctx accumulator (needed every call).
// ---------------------------------------------------------------------------
__global__ void k_convert_weights(const float* __restrict__ wqkv,
                                  const float* __restrict__ wout,
                                  _Float16* __restrict__ wqkv_h,
                                  _Float16* __restrict__ wout_h,
                                  float* __restrict__ ctx) {
  int t = blockIdx.x * blockDim.x + threadIdx.x;
  const int NF1 = (QR_ / 16) * (C_ / 32) * 32;   // 24*8*32 = 6144 lane-frags
  const int NF2 = (C_ / 16) * (HID_ / 32) * 32;  // 16*4*32 = 2048 lane-frags
  if (t < NF1) {
    int lane = t & 31, tile = t >> 5;
    int kb = tile & 7, i = tile >> 3;            // 8 ktiles of C
    _Float16* dst = wqkv_h + (size_t)t * 16;
    for (int j = 0; j < 16; ++j) {
      int m, k; afrag_mk(lane, j, m, k);
      dst[j] = (_Float16)wqkv[(i * 16 + m) * C_ + kb * 32 + k];
    }
  } else if (t < NF1 + NF2) {
    int t2 = t - NF1;
    int lane = t2 & 31, tile = t2 >> 5;
    int kb = tile & 3, i = tile >> 2;            // 4 ktiles of HID
    _Float16* dst = wout_h + (size_t)t2 * 16;
    for (int j = 0; j < 16; ++j) {
      int m, k; afrag_mk(lane, j, m, k);
      dst[j] = (_Float16)wout[(i * 16 + m) * HID_ + kb * 32 + k];
    }
  }
  if (t < B_ * NH_ * HD_ * HD_) ctx[t] = 0.f;    // 65536 floats
}

// ---------------------------------------------------------------------------
// K1: fused channel-RMSnorm + QKV GEMM (384x256 @ 256x64 per block).
// Output: f16 token-major qkvT[b][n][row] (16-byte v8h stores per tile).
// ---------------------------------------------------------------------------
__global__ __launch_bounds__(256) void k_qkv(const float* __restrict__ x,
                                             const float* __restrict__ g_in,
                                             const _Float16* __restrict__ wqkv_h,
                                             _Float16* __restrict__ qkvT) {
  __shared__ __align__(32) _Float16 xnT[64 * C_];  // [p][c] f16 B-frags, 32 KB
  __shared__ float s_part[4][64];
  __shared__ float s_rs[64];

  const int b = blockIdx.y;
  const int n0 = blockIdx.x * 64;
  const int tid = threadIdx.x;
  const int p = tid & 63, cg = tid >> 6;

  const float* xb = x + (size_t)b * C_ * N_ + n0 + p;

  float ss = 0.f;
  for (int c = cg * 64; c < cg * 64 + 64; ++c) {
    float v = xb[(size_t)c * N_];
    ss += v * v;
  }
  s_part[cg][p] = ss;
  __syncthreads();
  if (cg == 0) {
    float tot = s_part[0][p] + s_part[1][p] + s_part[2][p] + s_part[3][p];
    s_rs[p] = 16.0f / fmaxf(sqrtf(tot), 1e-12f);  // sqrt(C)=16
  }
  __syncthreads();

  float rs = s_rs[p];
  for (int c = cg * 64; c < cg * 64 + 64; ++c)
    xnT[p * C_ + c] = (_Float16)(xb[(size_t)c * N_] * rs * g_in[c]);
  __syncthreads();

  // GEMM: 24 row-tiles x 4 col-tiles of 16x16; wave w owns rowtiles 3w..3w+2
  const int wave = tid >> 5, lane = tid & 31, lcol = lane & 15;
  const int koff = (lane < 16) ? 0 : 16;
  v8f acc[3][4];
  for (int rr = 0; rr < 3; ++rr)
    for (int nt = 0; nt < 4; ++nt) acc[rr][nt] = vzero8();

  // A-fragment base pointers (kb-stride 512 halfs -> immediate offsets)
  const _Float16* aptr[3];
  for (int rr = 0; rr < 3; ++rr)
    aptr[rr] = wqkv_h + ((size_t)(wave * 3 + rr) * 8 * 32 + lane) * 16;

  for (int kb = 0; kb < 8; ++kb) {
    v16h bf[4];
    for (int nt = 0; nt < 4; ++nt)
      bf[nt] = *(const v16h*)&xnT[(nt * 16 + lcol) * C_ + kb * 32 + koff];
    for (int rr = 0; rr < 3; ++rr) {
      v16h af = *(const v16h*)(aptr[rr] + kb * 512);
      for (int nt = 0; nt < 4; ++nt)
        acc[rr][nt] = __builtin_amdgcn_wmma_f32_16x16x32_f16(
            false, af, false, bf[nt], (short)0, acc[rr][nt], false, false);
    }
  }

  // store f16 token-major: rows contiguous -> one 16B store per tile per lane
  const int rowadd = (lane >= 16) ? 8 : 0;
  for (int rr = 0; rr < 3; ++rr)
    for (int nt = 0; nt < 4; ++nt) {
      int rt = wave * 3 + rr;
      v8h hv;
      for (int j = 0; j < 8; ++j) hv[j] = (_Float16)acc[rr][nt][j];
      int col = n0 + nt * 16 + lcol;
      *(v8h*)(qkvT + ((size_t)b * N_ + col) * QR_ + rt * 16 + rowadd) = hv;
    }
}

// ---------------------------------------------------------------------------
// K2a: single-pass online k-softmax stats (max, sum-exp) per (b,h,d) row.
// Lane = head-dim channel -> 64B coalesced reads; k read exactly once.
// ---------------------------------------------------------------------------
__global__ __launch_bounds__(256) void k_kstats(const _Float16* __restrict__ qkvT,
                                                const float* __restrict__ mem_kv,
                                                float* __restrict__ stats) {
  __shared__ float pm[8][32], ps[8][32];
  const int h = blockIdx.x, b = blockIdx.y;
  const int tid = threadIdx.x, d = tid & 31, tg = tid >> 5;

  const _Float16* kp =
      qkvT + ((size_t)b * N_ + tg * 512) * QR_ + HID_ + h * 32 + d;
  float m = -3.4e38f, s = 0.f;
  for (int t = 0; t < 512; ++t) {
    float x = (float)kp[(size_t)t * QR_];
    float mn = fmaxf(m, x);
    s = s * __expf(m - mn) + __expf(x - mn);
    m = mn;
  }
  pm[tg][d] = m;
  ps[tg][d] = s;
  __syncthreads();
  if (tid < 32) {
    float M = -3.4e38f, S = 0.f;
    for (int i = 0; i < 8; ++i) {
      float mi = pm[i][tid], si = ps[i][tid];
      float Mn = fmaxf(M, mi);
      S = S * __expf(M - Mn) + si * __expf(mi - Mn);
      M = Mn;
    }
    for (int mt = 0; mt < NM_; ++mt) {  // memory tokens
      float xv = mem_kv[((0 * NH_ + h) * HD_ + tid) * NM_ + mt];
      float Mn = fmaxf(M, xv);
      S = S * __expf(M - Mn) + __expf(xv - Mn);
      M = Mn;
    }
    stats[(b * 128 + h * 32 + tid) * 2 + 0] = M;
    stats[(b * 128 + h * 32 + tid) * 2 + 1] = S;
  }
}

// ---------------------------------------------------------------------------
// K2b: unnormalized context += sum_tokens exp(k-M)[d,n] * v[e,n].
// Grid (chunk=8, NH, B) = 512 blocks; f32 atomics into ctx; 1/S folded in K3.
// ---------------------------------------------------------------------------
__global__ __launch_bounds__(256) void k_context(const _Float16* __restrict__ qkvT,
                                                 const float* __restrict__ mem_kv,
                                                 const float* __restrict__ stats,
                                                 float* __restrict__ ctx) {
  __shared__ float kt[128][33];   // padded: conflict-free staging
  __shared__ float vt[128][33];
  __shared__ float sM[32];
  const int chunk = blockIdx.x, h = blockIdx.y, b = blockIdx.z;
  const int tid = threadIdx.x;
  if (tid < 32) sM[tid] = stats[(b * 128 + h * 32 + tid) * 2];
  __syncthreads();

  const int d = tid >> 3, eg = tid & 7;  // thread owns (d, e = eg*4..eg*4+3)
  float acc[4] = {0.f, 0.f, 0.f, 0.f};

  if (chunk == 0) {  // memory tokens (unnormalized)
    for (int i = 0; i < 4; ++i)
      for (int mt = 0; mt < NM_; ++mt)
        acc[i] += __expf(mem_kv[((0 * NH_ + h) * HD_ + d) * NM_ + mt] - sM[d]) *
                  mem_kv[((1 * NH_ + h) * HD_ + (eg * 4 + i)) * NM_ + mt];
  }

  const int t0 = tid >> 1, half = tid & 1;  // two threads stage one token
  for (int sc = 0; sc < 4; ++sc) {
    int tokbase = chunk * 512 + sc * 128;
    const _Float16* base = qkvT + ((size_t)b * N_ + tokbase + t0) * QR_;
    v16h kv16 = *(const v16h*)(base + HID_ + h * 32 + half * 16);
    v16h vv16 = *(const v16h*)(base + 2 * HID_ + h * 32 + half * 16);
    for (int j = 0; j < 16; ++j) {
      int dd = half * 16 + j;
      kt[t0][dd] = __expf((float)kv16[j] - sM[dd]);
      vt[t0][dd] = (float)vv16[j];
    }
    __syncthreads();
    for (int t = 0; t < 128; ++t) {
      float kvv = kt[t][d];
      for (int i = 0; i < 4; ++i) acc[i] += kvv * vt[t][eg * 4 + i];
    }
    __syncthreads();
  }

  for (int i = 0; i < 4; ++i)
    atomicAdd(&ctx[(((size_t)b * NH_ + h) * HD_ + d) * HD_ + eg * 4 + i],
              acc[i]);
}

// ---------------------------------------------------------------------------
// K3: fused tail per 64-pixel tile:
//   q-softmax*scale -> WMMA ctx^T @ q_sm -> WMMA w_out @ attn -> +bias
//   -> channel RMS-norm (ds_add_f32 cross-wave ssq) -> single output store.
// ---------------------------------------------------------------------------
__global__ __launch_bounds__(256) void k_output(const _Float16* __restrict__ qkvT,
                                                const float* __restrict__ ctx,
                                                const float* __restrict__ stats,
                                                const _Float16* __restrict__ wout_h,
                                                const float* __restrict__ b_out,
                                                const float* __restrict__ g_out,
                                                float* __restrict__ out) {
  __shared__ __align__(32) _Float16 ctxA[4 * 2 * 32 * 16];  // A-frags, 8 KB
  __shared__ __align__(32) _Float16 qT[4 * 64 * 32];        // [h][p][d], 16 KB
  __shared__ __align__(32) _Float16 attnT[64 * 128];        // [p][o],    16 KB
  __shared__ float ssq[64];

  const int b = blockIdx.y;
  const int n0 = blockIdx.x * 64;
  const int tid = threadIdx.x;

  // A-fragments of normalized ctx^T:  A[e][d] = ctx[d][e] / S[d]
  {
    int lane = tid & 31, et = (tid >> 5) & 1, h = tid >> 6;
    _Float16* dst = &ctxA[(((h * 2) + et) * 32 + lane) * 16];
    const float* cb = ctx + ((size_t)b * NH_ + h) * HD_ * HD_;
    const float* st = stats + (b * 128 + h * 32) * 2;
    for (int j = 0; j < 16; ++j) {
      int m, k; afrag_mk(lane, j, m, k);
      dst[j] = (_Float16)(cb[k * HD_ + et * 16 + m] / st[k * 2 + 1]);
    }
  }

  // q softmax over head-dim, times HD^-0.5; token-major q -> two 32-byte v16h loads
  {
    int p = tid & 63, h = tid >> 6;
    const _Float16* qp = qkvT + ((size_t)b * N_ + n0 + p) * QR_ + h * 32;
    v16h q0 = *(const v16h*)qp;
    v16h q1 = *(const v16h*)(qp + 16);
    float vals[32];
    float mx = -3.4e38f;
    for (int d = 0; d < 16; ++d) { vals[d] = (float)q0[d]; mx = fmaxf(mx, vals[d]); }
    for (int d = 0; d < 16; ++d) { vals[16 + d] = (float)q1[d]; mx = fmaxf(mx, vals[16 + d]); }
    float sm = 0.f;
    for (int d = 0; d < 32; ++d) { vals[d] = __expf(vals[d] - mx); sm += vals[d]; }
    float inv = 0.17677669529663687f / sm;  // HD^-0.5 / sum
    for (int d = 0; d < 32; ++d)
      qT[(h * 64 + p) * 32 + d] = (_Float16)(vals[d] * inv);
  }
  if (tid < 64) ssq[tid] = 0.f;
  __syncthreads();

  const int wave = tid >> 5, lane = tid & 31, lcol = lane & 15;
  const int koff = (lane < 16) ? 0 : 16;
  const int rowadd = (lane >= 16) ? 8 : 0;

  // GEMM1: attn[o][p] = ctx^T @ q_sm. Wave -> (head = w>>1, etile = w&1).
  {
    int h = wave >> 1, et = wave & 1;
    v16h af = *(const v16h*)&ctxA[(((h * 2) + et) * 32 + lane) * 16];
    for (int nt = 0; nt < 4; ++nt) {
      v16h bf = *(const v16h*)&qT[(h * 64 + nt * 16 + lcol) * 32 + koff];
      v8f a = vzero8();
      a = __builtin_amdgcn_wmma_f32_16x16x32_f16(false, af, false, bf,
                                                 (short)0, a, false, false);
      for (int j = 0; j < 8; ++j) {
        int o = h * 32 + et * 16 + j + rowadd;
        attnT[(nt * 16 + lcol) * 128 + o] = (_Float16)a[j];
      }
    }
  }
  __syncthreads();

  // GEMM2: y[c][p] = w_out @ attn. Wave -> rowtiles {2w, 2w+1} of C.
  v8f acc2[2][4];
  for (int rr = 0; rr < 2; ++rr)
    for (int nt = 0; nt < 4; ++nt) acc2[rr][nt] = vzero8();

  for (int kb = 0; kb < 4; ++kb) {
    v16h bf[4];
    for (int nt = 0; nt < 4; ++nt)
      bf[nt] = *(const v16h*)&attnT[(nt * 16 + lcol) * 128 + kb * 32 + koff];
    for (int rr = 0; rr < 2; ++rr) {
      int rt = wave * 2 + rr;
      v16h af = *(const v16h*)(wout_h + (((size_t)rt * 4 + kb) * 32 + lane) * 16);
      for (int nt = 0; nt < 4; ++nt)
        acc2[rr][nt] = __builtin_amdgcn_wmma_f32_16x16x32_f16(
            false, af, false, bf[nt], (short)0, acc2[rr][nt], false, false);
    }
  }

  // bias + cross-channel sum-of-squares (ds_add_f32)
  for (int rr = 0; rr < 2; ++rr)
    for (int nt = 0; nt < 4; ++nt) {
      int rt = wave * 2 + rr;
      float s = 0.f;
      for (int j = 0; j < 8; ++j) {
        int c = rt * 16 + j + rowadd;
        float v = acc2[rr][nt][j] + b_out[c];
        acc2[rr][nt][j] = v;
        s += v * v;
      }
      atomicAdd(&ssq[nt * 16 + lcol], s);
    }
  __syncthreads();

  // RMS-norm over C=256 and final store
  for (int rr = 0; rr < 2; ++rr)
    for (int nt = 0; nt < 4; ++nt) {
      int rt = wave * 2 + rr;
      float rinv = 16.0f / fmaxf(sqrtf(ssq[nt * 16 + lcol]), 1e-12f);
      for (int j = 0; j < 8; ++j) {
        int c = rt * 16 + j + rowadd;
        out[((size_t)b * C_ + c) * N_ + n0 + nt * 16 + lcol] =
            acc2[rr][nt][j] * rinv * g_out[c];
      }
    }
}

// ---------------------------------------------------------------------------
// Host side
// ---------------------------------------------------------------------------
extern "C" void kernel_launch(void* const* d_in, const int* in_sizes, int n_in,
                              void* d_out, int out_size, void* d_ws, size_t ws_size,
                              hipStream_t stream) {
  (void)in_sizes; (void)n_in; (void)out_size; (void)ws_size;
  const float* x     = (const float*)d_in[0];
  const float* g_in  = (const float*)d_in[1];
  const float* memkv = (const float*)d_in[2];
  const float* wqkv  = (const float*)d_in[3];
  const float* wout  = (const float*)d_in[4];
  const float* bout  = (const float*)d_in[5];
  const float* gout  = (const float*)d_in[6];
  float* out = (float*)d_out;

  char* ws = (char*)d_ws;
  size_t off = 0;
  _Float16* qkvT = (_Float16*)(ws + off); off += (size_t)B_ * N_ * QR_ * 2;   // 48 MiB
  float* stats = (float*)(ws + off);      off += (size_t)B_ * 128 * 2 * 4;    // 16 KB
  float* ctx = (float*)(ws + off);        off += (size_t)B_ * NH_ * HD_ * HD_ * 4; // 256 KB
  _Float16* wqkv_h = (_Float16*)(ws + off); off += (size_t)QR_ * C_ * 2;      // 192 KB
  _Float16* wout_h = (_Float16*)(ws + off); off += (size_t)C_ * HID_ * 2;     // 64 KB

  // K0: weight swizzle + ctx zeroing (65536 threads)
  k_convert_weights<<<256, 256, 0, stream>>>(wqkv, wout, wqkv_h, wout_h, ctx);

  dim3 gtile(N_ / 64, B_);
  k_qkv<<<gtile, 256, 0, stream>>>(x, g_in, wqkv_h, qkvT);
  k_kstats<<<dim3(NH_, B_), 256, 0, stream>>>(qkvT, memkv, stats);
  k_context<<<dim3(8, NH_, B_), 256, 0, stream>>>(qkvT, memkv, stats, ctx);
  k_output<<<gtile, 256, 0, stream>>>(qkvT, ctx, stats, wout_h, bout, gout, out);
}
